// BPModel_32272384262812
// MI455X (gfx1250) — matrix operand order repaired and checked by reading, not compiled, not run on hardware
//
#include <hip/hip_runtime.h>
#include <math.h>

// ---------------- problem constants ----------------
#define VSZ   50000
#define ESZ   256
#define HSZ   128
#define TSZ   64
#define LSZ   4096
#define DLVL  12
#define NNODE 8191            // 2*L-1
#define GSZ   512             // 4*H
#define TWOH  256
#define TT    4096            // T*T

// edge GEMM tiling
#define EMT   512             // M tiles (8192 padded rows / 16)
#define ENT   256             // N tiles (4096 / 16)
#define EKC   8               // K chunks (256 / 32)

// ---------------- WMMA fragment types ----------------
typedef __attribute__((ext_vector_type(16))) _Float16 v16h;
typedef __attribute__((ext_vector_type(8)))  float    v8f;

union AF { v16h v; _Float16 e[16]; };
union CF { v8f  v; float    e[8]; };

// 16-bit A (16x32): lane m=l&15, h=l>>4; VGPR v holds K = (v>=4?16:0)+8h+2(v&3), +1
__device__ __forceinline__ int a_kidx(int v, int h) {
    return ((v & 4) << 2) + (h << 3) + ((v & 3) << 1);
}
// 16-bit B (32x16): lane n=l&15, h=l>>4; VGPR v holds K = (v>=4?16:0)+8h+2(v&3), +1
__device__ __forceinline__ int b_kidx(int v, int h) {
    return ((v & 4) << 2) + (h << 3) + ((v & 3) << 1);
}

__device__ __forceinline__ float sigm(float x) { return 1.0f / (1.0f + expf(-x)); }

// =====================================================================
// K1: xW[l,g] = (emb[tokens[l]] @ W_ih^T)[g] + b_ih[g] + b_hh[g]
//     M=4096, N=512, K=256.  8192 tiles, 1 wave per 16x16 tile.
// =====================================================================
__global__ void gemm_xw_kernel(const int* __restrict__ tokens,
                               const float* __restrict__ emb,
                               const float* __restrict__ W_ih,
                               const float* __restrict__ b_ih,
                               const float* __restrict__ b_hh,
                               float* __restrict__ xW)
{
    int wave = (blockIdx.x * blockDim.x + threadIdx.x) >> 5;
    if (wave >= 256 * 32) return;
    int mt = wave >> 5, nt = wave & 31;
    int lane = threadIdx.x & 31;
    int mr = lane & 15, h = lane >> 4;

    int row  = mt * 16 + mr;
    const float* arow = emb + (size_t)tokens[row] * ESZ;
    int ncol = nt * 16 + mr;
    const float* bcol = W_ih + (size_t)ncol * ESZ;    // B[k][n] = W_ih[n][k]

    v8f acc = {};
    for (int k0 = 0; k0 < ESZ; k0 += 32) {
        AF a; AF b;
        #pragma unroll
        for (int v = 0; v < 8; ++v) {
            int ka = k0 + a_kidx(v, h);
            a.e[2*v]   = (_Float16)arow[ka];
            a.e[2*v+1] = (_Float16)arow[ka + 1];
            int kb = k0 + b_kidx(v, h);
            b.e[2*v]   = (_Float16)bcol[kb];
            b.e[2*v+1] = (_Float16)bcol[kb + 1];
        }
        acc = __builtin_amdgcn_wmma_f32_16x16x32_f16(false, a.v, false, b.v,
                                                     (short)0, acc, false, false);
    }
    float bias = b_ih[ncol] + b_hh[ncol];
    CF c; c.v = acc;
    #pragma unroll
    for (int r = 0; r < 8; ++r)
        xW[(size_t)(mt * 16 + r + 8 * h) * GSZ + ncol] = c.e[r] + bias;
}

// =====================================================================
// K2: sequential LSTM over 4096 steps.  One block, 1024 threads:
//     thread = 2*g + half, each owns 64 W_hh weights in registers.
// =====================================================================
__global__ void __launch_bounds__(1024) lstm_kernel(const float* __restrict__ xW,
                                                    const float* __restrict__ W_hh,
                                                    float* __restrict__ hidden)
{
    __shared__ float h_s[HSZ];
    __shared__ float c_s[HSZ];
    __shared__ float red[1024];
    __shared__ float gpre[GSZ];

    int tid  = threadIdx.x;
    int g    = tid >> 1;
    int half = tid & 1;

    float w[64];
    const float* wrow = W_hh + (size_t)g * HSZ + half * 64;
    #pragma unroll
    for (int k = 0; k < 64; ++k) w[k] = wrow[k];

    if (tid < HSZ) { h_s[tid] = 0.0f; c_s[tid] = 0.0f; }
    __syncthreads();

    for (int t = 0; t < LSZ; ++t) {
        float acc = 0.0f;
        #pragma unroll
        for (int k = 0; k < 64; ++k) acc += w[k] * h_s[half * 64 + k];
        red[tid] = acc;
        __syncthreads();
        if (tid < GSZ)
            gpre[tid] = xW[(size_t)t * GSZ + tid] + red[2 * tid] + red[2 * tid + 1];
        __syncthreads();
        if (tid < HSZ) {
            float iv = gpre[tid];
            float fv = gpre[HSZ + tid];
            float gv = gpre[2 * HSZ + tid];
            float ov = gpre[3 * HSZ + tid];
            float cc = sigm(fv) * c_s[tid] + sigm(iv) * tanhf(gv);
            float hh = sigm(ov) * tanhf(cc);
            c_s[tid] = cc;
            h_s[tid] = hh;
            hidden[(size_t)(LSZ - 1 + t) * HSZ + tid] = hh;
        }
        __syncthreads();
    }
}

// =====================================================================
// K3: one tree level: hidden[idx] = [hidden[2i+1];hidden[2i+2]] @ W_p2h^T + b
// =====================================================================
__global__ void p2h_kernel(const float* __restrict__ W_p2h,
                           const float* __restrict__ b_p2h,
                           float* __restrict__ hidden,
                           int idx0, int cnt)
{
    int mtiles = (cnt + 15) >> 4;
    int wave = (blockIdx.x * blockDim.x + threadIdx.x) >> 5;
    if (wave >= mtiles * 8) return;
    int mt = wave >> 3, nt = wave & 7;
    int lane = threadIdx.x & 31;
    int mr = lane & 15, h = lane >> 4;

    int mloc = mt * 16 + mr;
    int node = idx0 + (mloc < cnt ? mloc : cnt - 1);   // clamp; stores are guarded
    const float* lrow = hidden + (size_t)(2 * node + 1) * HSZ;
    const float* rrow = hidden + (size_t)(2 * node + 2) * HSZ;
    int ncol = nt * 16 + mr;
    const float* bcol = W_p2h + (size_t)ncol * TWOH;   // B[k][n] = W_p2h[n][k]

    v8f acc = {};
    for (int k0 = 0; k0 < TWOH; k0 += 32) {
        AF a; AF b;
        #pragma unroll
        for (int v = 0; v < 8; ++v) {
            int ka = k0 + a_kidx(v, h);
            float x0 = (ka     < HSZ) ? lrow[ka]     : rrow[ka - HSZ];
            float x1 = (ka + 1 < HSZ) ? lrow[ka + 1] : rrow[ka + 1 - HSZ];
            a.e[2*v]   = (_Float16)x0;
            a.e[2*v+1] = (_Float16)x1;
            int kb = k0 + b_kidx(v, h);
            b.e[2*v]   = (_Float16)bcol[kb];
            b.e[2*v+1] = (_Float16)bcol[kb + 1];
        }
        acc = __builtin_amdgcn_wmma_f32_16x16x32_f16(false, a.v, false, b.v,
                                                     (short)0, acc, false, false);
    }
    CF c; c.v = acc;
    #pragma unroll
    for (int r = 0; r < 8; ++r) {
        int mm = mt * 16 + r + 8 * h;
        if (mm < cnt)
            hidden[(size_t)(idx0 + mm) * HSZ + ncol] = c.e[r] + b_p2h[ncol];
    }
}

// =====================================================================
// K4: unary = hidden @ W_uni^T + b_uni.  M=8191, N=64, K=128.
// =====================================================================
__global__ void unary_kernel(const float* __restrict__ W_uni,
                             const float* __restrict__ b_uni,
                             const float* __restrict__ hidden,
                             float* __restrict__ unary)
{
    int wave = (blockIdx.x * blockDim.x + threadIdx.x) >> 5;
    if (wave >= 512 * 4) return;
    int mt = wave >> 2, nt = wave & 3;
    int lane = threadIdx.x & 31;
    int mr = lane & 15, h = lane >> 4;

    int row = mt * 16 + mr;
    const float* arow = hidden + (size_t)(row < NNODE ? row : NNODE - 1) * HSZ;
    int ncol = nt * 16 + mr;
    const float* bcol = W_uni + (size_t)ncol * HSZ;

    v8f acc = {};
    for (int k0 = 0; k0 < HSZ; k0 += 32) {
        AF a; AF b;
        #pragma unroll
        for (int v = 0; v < 8; ++v) {
            int ka = k0 + a_kidx(v, h);
            a.e[2*v]   = (_Float16)arow[ka];
            a.e[2*v+1] = (_Float16)arow[ka + 1];
            int kb = k0 + b_kidx(v, h);
            b.e[2*v]   = (_Float16)bcol[kb];
            b.e[2*v+1] = (_Float16)bcol[kb + 1];
        }
        acc = __builtin_amdgcn_wmma_f32_16x16x32_f16(false, a.v, false, b.v,
                                                     (short)0, acc, false, false);
    }
    CF c; c.v = acc;
    #pragma unroll
    for (int r = 0; r < 8; ++r) {
        int mm = mt * 16 + r + 8 * h;
        if (mm < NNODE)
            unary[(size_t)mm * TSZ + ncol] = c.e[r] + b_uni[ncol];
    }
}

// =====================================================================
// K5a: pack W_edge^T into f16 WMMA B-fragment layout:
//      bpk[((nt*EKC + kc)*32 + lane)*16 + j]
// =====================================================================
__global__ void pack_b_edge_kernel(const float* __restrict__ W_edge,
                                   _Float16* __restrict__ bpk)
{
    int gid = blockIdx.x * blockDim.x + threadIdx.x;
    if (gid >= ENT * EKC * 32) return;
    int lane = gid & 31;
    int kc   = (gid >> 5) & (EKC - 1);
    int nt   = gid >> 8;
    int h    = lane >> 4;
    int n    = nt * 16 + (lane & 15);

    const float* bcol = W_edge + (size_t)n * TWOH + kc * 32;
    _Float16* out = bpk + (size_t)gid * 16;
    #pragma unroll
    for (int v = 0; v < 8; ++v) {
        int kb = b_kidx(v, h);
        out[2*v]   = (_Float16)bcol[kb];
        out[2*v+1] = (_Float16)bcol[kb + 1];
    }
}

// =====================================================================
// K5b: pack pair matrix [hidden[parent];hidden[child]] (rows = child-1)
//      into f16 WMMA A-fragment layout: apk[((mt*EKC + kc)*32 + lane)*16 + j]
//      Rows >= 8190 are zero-filled.
// =====================================================================
__global__ void pack_a_edge_kernel(const float* __restrict__ hidden,
                                   _Float16* __restrict__ apk)
{
    int gid = blockIdx.x * blockDim.x + threadIdx.x;
    if (gid >= EMT * EKC * 32) return;
    int lane = gid & 31;
    int kc   = (gid >> 5) & (EKC - 1);
    int mt   = gid >> 8;
    int mr   = lane & 15, h = lane >> 4;

    int rowi = mt * 16 + mr;
    bool valid = rowi < NNODE - 1;
    int child  = (valid ? rowi : 0) + 1;
    int parent = (child - 1) >> 1;
    const float* prow = hidden + (size_t)parent * HSZ;
    const float* crow = hidden + (size_t)child * HSZ;

    _Float16* out = apk + (size_t)gid * 16;
    #pragma unroll
    for (int v = 0; v < 8; ++v) {
        int ka = kc * 32 + a_kidx(v, h);
        float x0 = valid ? ((ka     < HSZ) ? prow[ka]     : crow[ka - HSZ])     : 0.0f;
        float x1 = valid ? ((ka + 1 < HSZ) ? prow[ka + 1] : crow[ka + 1 - HSZ]) : 0.0f;
        out[2*v]   = (_Float16)x0;
        out[2*v+1] = (_Float16)x1;
    }
}

// =====================================================================
// K5c: edge factor GEMM from packed fragments.
//      M=8190(+pad), N=4096, K=256.  One wave computes a 16x64 strip
//      (1 A fragment reused across 4 N-tiles, 4 accumulators).
//      Inner loop: 5 contiguous 32B fragment loads -> 4 WMMAs.
// =====================================================================
__global__ void edge_gemm_kernel(const _Float16* __restrict__ apk,
                                 const _Float16* __restrict__ bpk,
                                 const float* __restrict__ b_edge,
                                 float* __restrict__ edge)
{
    int wave = (blockIdx.x * blockDim.x + threadIdx.x) >> 5;
    if (wave >= EMT * (ENT / 4)) return;
    int mt = wave >> 6;              // M tile
    int ng = wave & 63;              // group of 4 N tiles
    int lane = threadIdx.x & 31;
    int mr = lane & 15, h = lane >> 4;

    const v16h* ap = (const v16h*)apk + (size_t)mt * EKC * 32 + lane;
    const v16h* bp = (const v16h*)bpk + (size_t)(ng * 4) * EKC * 32 + lane;

    // prefetch next M tile's A pack (lowers to global_prefetch_b8)
    __builtin_prefetch((const void*)(ap + EKC * 32), 0, 0);

    v8f acc0 = {}, acc1 = {}, acc2 = {}, acc3 = {};
    #pragma unroll
    for (int kc = 0; kc < EKC; ++kc) {
        AF a;              a.v  = ap[kc * 32];
        AF b0, b1, b2, b3;
        b0.v = bp[(0 * EKC + kc) * 32];
        b1.v = bp[(1 * EKC + kc) * 32];
        b2.v = bp[(2 * EKC + kc) * 32];
        b3.v = bp[(3 * EKC + kc) * 32];
        acc0 = __builtin_amdgcn_wmma_f32_16x16x32_f16(false, a.v, false, b0.v,
                                                      (short)0, acc0, false, false);
        acc1 = __builtin_amdgcn_wmma_f32_16x16x32_f16(false, a.v, false, b1.v,
                                                      (short)0, acc1, false, false);
        acc2 = __builtin_amdgcn_wmma_f32_16x16x32_f16(false, a.v, false, b2.v,
                                                      (short)0, acc2, false, false);
        acc3 = __builtin_amdgcn_wmma_f32_16x16x32_f16(false, a.v, false, b3.v,
                                                      (short)0, acc3, false, false);
    }

    CF c[4];
    c[0].v = acc0; c[1].v = acc1; c[2].v = acc2; c[3].v = acc3;
    #pragma unroll
    for (int j = 0; j < 4; ++j) {
        int ncol = ng * 64 + j * 16 + mr;
        float bias = b_edge[ncol];
        #pragma unroll
        for (int r = 0; r < 8; ++r) {
            int mm = mt * 16 + r + 8 * h;
            if (mm < NNODE - 1)
                edge[(size_t)mm * TT + ncol] = c[j].e[r] + bias;
        }
    }
}

// =====================================================================
// K6: init v2f = unary, f2n = 0
// =====================================================================
__global__ void init_msgs_kernel(const float* __restrict__ unary,
                                 float* __restrict__ v2f,
                                 float* __restrict__ f2n)
{
    int gid = blockIdx.x * blockDim.x + threadIdx.x;
    if (gid < NNODE * TSZ) {
        v2f[gid] = unary[gid];
        f2n[gid] = 0.0f;
    }
}

// =====================================================================
// K7: upward level: v2f[idx] += f2p[children]; f2p[idx][t] =
//     logsumexp_c(edge[idx-1][t][c] + v2f[idx][c]).  64 thr/node.
// =====================================================================
__global__ void up_level_kernel(const float* __restrict__ edge,
                                float* __restrict__ v2f,
                                float* __restrict__ f2p,
                                int idx0, int cnt, int addChildren, int emitParent)
{
    __shared__ float vrow[4][TSZ];
    int nl = threadIdx.x >> 6;
    int t  = threadIdx.x & 63;
    int ni = blockIdx.x * 4 + nl;
    bool valid = ni < cnt;
    int node = idx0 + (valid ? ni : 0);

    float val = 0.0f;
    if (valid) {
        val = v2f[(size_t)node * TSZ + t];
        if (addChildren)
            val += f2p[(size_t)(2 * node + 1) * TSZ + t]
                 + f2p[(size_t)(2 * node + 2) * TSZ + t];
        v2f[(size_t)node * TSZ + t] = val;
    }
    vrow[nl][t] = val;
    __syncthreads();

    if (emitParent && valid) {
        const float* er = edge + (size_t)(node - 1) * TT + t * TSZ;
        float ev[TSZ];
        float mx = -3.4e38f;
        #pragma unroll
        for (int c = 0; c < TSZ; ++c) {
            ev[c] = er[c] + vrow[nl][c];
            mx = fmaxf(mx, ev[c]);
        }
        float s = 0.0f;
        #pragma unroll
        for (int c = 0; c < TSZ; ++c) s += expf(ev[c] - mx);
        f2p[(size_t)node * TSZ + t] = mx + logf(s);
    }
}

// =====================================================================
// K8: downward level: p2f = unary[par] + f2n[par] + f2p[sib];
//     f2n[idx][c] = logsumexp_t(edge[idx-1][t][c] + p2f[t]).
// =====================================================================
__global__ void down_level_kernel(const float* __restrict__ edge,
                                  const float* __restrict__ unary,
                                  const float* __restrict__ f2p,
                                  float* __restrict__ f2n,
                                  int idx0, int cnt)
{
    __shared__ float prow[4][TSZ];
    int nl = threadIdx.x >> 6;
    int c  = threadIdx.x & 63;
    int ni = blockIdx.x * 4 + nl;
    bool valid = ni < cnt;
    int node = idx0 + (valid ? ni : 0);

    if (valid) {
        int par = (node - 1) >> 1;
        int sib = (node & 1) ? node + 1 : node - 1;
        prow[nl][c] = unary[(size_t)par * TSZ + c]
                    + f2n[(size_t)par * TSZ + c]
                    + f2p[(size_t)sib * TSZ + c];
    } else {
        prow[nl][c] = 0.0f;
    }
    __syncthreads();

    if (valid) {
        const float* ec = edge + (size_t)(node - 1) * TT + c;
        float ev[TSZ];
        float mx = -3.4e38f;
        #pragma unroll
        for (int t = 0; t < TSZ; ++t) {
            ev[t] = ec[t * TSZ] + prow[nl][t];
            mx = fmaxf(mx, ev[t]);
        }
        float s = 0.0f;
        #pragma unroll
        for (int t = 0; t < TSZ; ++t) s += expf(ev[t] - mx);
        f2n[(size_t)node * TSZ + c] = mx + logf(s);
    }
}

// =====================================================================
// K9: beliefs = v2f + f2n, scattered to postorder position computed in
//     closed form from the heap-index path bits of (node+1).
// =====================================================================
__global__ void beliefs_post_kernel(const float* __restrict__ v2f,
                                    const float* __restrict__ f2n,
                                    float* __restrict__ out)
{
    int gid = blockIdx.x * blockDim.x + threadIdx.x;
    if (gid >= NNODE * TSZ) return;
    int node = gid >> 6;
    int t    = gid & 63;

    unsigned x = (unsigned)(node + 1);
    int depth = 31 - __clz((int)x);
    int m = NNODE, pos = 0;
    for (int b = depth - 1; b >= 0; --b) {
        int s = (m - 1) >> 1;
        if ((x >> b) & 1u) pos += s;   // went right: skip left subtree
        m = s;
    }
    pos += m - 1;                       // node is last within its own subtree

    out[(size_t)pos * TSZ + t] = v2f[gid] + f2n[gid];
}

// =====================================================================
// host launcher
// =====================================================================
extern "C" void kernel_launch(void* const* d_in, const int* in_sizes, int n_in,
                              void* d_out, int out_size, void* d_ws, size_t ws_size,
                              hipStream_t stream)
{
    (void)in_sizes; (void)n_in; (void)out_size; (void)ws_size;

    const int*   tokens = (const int*)  d_in[0];
    const float* emb    = (const float*)d_in[1];
    const float* W_ih   = (const float*)d_in[2];
    const float* W_hh   = (const float*)d_in[3];
    const float* b_ih   = (const float*)d_in[4];
    const float* b_hh   = (const float*)d_in[5];
    const float* W_p2h  = (const float*)d_in[6];
    const float* b_p2h  = (const float*)d_in[7];
    const float* W_uni  = (const float*)d_in[8];
    const float* b_uni  = (const float*)d_in[9];
    const float* W_edge = (const float*)d_in[10];
    const float* b_edge = (const float*)d_in[11];
    float* out = (float*)d_out;

    // workspace layout (floats)
    float* ws     = (float*)d_ws;
    float* xW     = ws;                               // 4096*512   = 2,097,152
    float* hidden = xW + (size_t)LSZ * GSZ;           // 8191*128   = 1,048,448
    float* unary  = hidden + (size_t)NNODE * HSZ;     // 8191*64    =   524,224
    float* v2f    = unary + (size_t)NNODE * TSZ;
    float* f2p    = v2f   + (size_t)NNODE * TSZ;
    float* f2n    = f2p   + (size_t)NNODE * TSZ;
    float* edge   = f2n   + (size_t)NNODE * TSZ;      // 8190*4096  = 33,546,240
    // f16 packed fragment buffers (32B-aligned by construction)
    _Float16* apk = (_Float16*)(edge + (size_t)(NNODE - 1) * TT);  // 512*8*32*16 = 2,097,152 halves
    _Float16* bpk = apk + (size_t)EMT * EKC * 32 * 16;             // 256*8*32*16 = 1,048,576 halves

    // 1) embedding gather + input projection GEMM (WMMA)
    gemm_xw_kernel<<<1024, 256, 0, stream>>>(tokens, emb, W_ih, b_ih, b_hh, xW);

    // pack W_edge while the LSTM runs behind it in stream order
    pack_b_edge_kernel<<<(ENT * EKC * 32) / 256, 256, 0, stream>>>(W_edge, bpk);

    // 2) sequential LSTM recurrence (single persistent block)
    lstm_kernel<<<1, 1024, 0, stream>>>(xW, W_hh, hidden);

    // 3) bottom-up pair->hidden tree levels (WMMA per level)
    for (int d = DLVL - 1; d >= 0; --d) {
        int idx0 = (1 << d) - 1, cnt = 1 << d;
        int waves = ((cnt + 15) / 16) * 8;
        int blocks = (waves * 32 + 255) / 256;
        p2h_kernel<<<blocks, 256, 0, stream>>>(W_p2h, b_p2h, hidden, idx0, cnt);
    }

    // 4) unary GEMM + packed edge GEMM (WMMA)
    unary_kernel<<<256, 256, 0, stream>>>(W_uni, b_uni, hidden, unary);
    pack_a_edge_kernel<<<(EMT * EKC * 32) / 256, 256, 0, stream>>>(hidden, apk);
    edge_gemm_kernel<<<(EMT * (ENT / 4)) * 32 / 256, 256, 0, stream>>>(apk, bpk, b_edge, edge);

    // 5) message init
    init_msgs_kernel<<<2048, 256, 0, stream>>>(unary, v2f, f2n);

    // 6) upward pass, level-synchronous
    for (int d = DLVL; d >= 0; --d) {
        int idx0 = (1 << d) - 1, cnt = 1 << d;
        up_level_kernel<<<(cnt + 3) / 4, 256, 0, stream>>>(edge, v2f, f2p,
                                                           idx0, cnt,
                                                           d < DLVL ? 1 : 0,
                                                           d > 0 ? 1 : 0);
    }

    // 7) downward pass, level-synchronous
    for (int d = 1; d <= DLVL; ++d) {
        int idx0 = (1 << d) - 1, cnt = 1 << d;
        down_level_kernel<<<(cnt + 3) / 4, 256, 0, stream>>>(edge, unary, f2p, f2n,
                                                             idx0, cnt);
    }

    // 8) beliefs + postorder scatter
    beliefs_post_kernel<<<2048, 256, 0, stream>>>(v2f, f2n, out);
}
